// P2BVoteNetRPN_63161789055461
// MI455X (gfx1250) — matrix operand order repaired and checked by reading, not compiled, not run on hardware
//
#include <hip/hip_runtime.h>

// ---------------------------------------------------------------------------
// P2B VoteNet RPN forward for MI455X (gfx1250, wave32, WMMA 16x16x32 f16).
// All GEMM-shaped work lowered to v_wmma_f32_16x16x32_f16 (f32 accumulate).
// ---------------------------------------------------------------------------

#define BB      128
#define NPTS    128
#define DD      256
#define KNB     16
#define NPROP   64
#define NSAMP   16
#define R2C     0.09f          // RADIUS^2
#define M1      (BB * NPTS)    // 16384 rows for per-point features
#define MSA     (BB * NPROP * NSAMP) // 131072
#define MP      (BB * NPROP)   // 8192

typedef __attribute__((ext_vector_type(16))) _Float16 v16h;
typedef __attribute__((ext_vector_type(8)))  float    v8f;

static __device__ __forceinline__ v8f wmma_f16(v16h a, v16h b, v8f c) {
    // (neg_a, A, neg_b, B, c_mod, C, reuse_a, reuse_b)
    return __builtin_amdgcn_wmma_f32_16x16x32_f16(false, a, false, b, (short)0, c,
                                                  false, false);
}

// ---------------------------------------------------------------------------
// Generic GEMM: C[M,N] = A[M,K] * W[N,K]^T (+bias)(+resid)(optional relu)
// A: fp32 row-major (lda), W: f16 row-major (ldw). K % 32 == 0, M % 64 == 0.
// Block = 128 threads (4 waves); each wave -> 16 rows x 64 cols (4 n-tiles).
// ---------------------------------------------------------------------------
__global__ __launch_bounds__(128)
void gemm_wmma_kernel(const float* __restrict__ A, int lda,
                      const _Float16* __restrict__ W, int ldw,
                      float* __restrict__ C, int ldc,
                      const float* __restrict__ bias,
                      const float* __restrict__ resid, int ldr,
                      int N, int K, int relu)
{
    const int wave = threadIdx.x >> 5;
    const int lane = threadIdx.x & 31;
    const int nl   = lane & 15;         // n within tile / m row for A frag
    const int hi   = lane >> 4;         // lane half selects K sub-block
    const int rowBase = blockIdx.x * 64 + wave * 16;
    const int colBase = blockIdx.y * 64;

    v8f acc[4];
#pragma unroll
    for (int t = 0; t < 4; ++t)
#pragma unroll
        for (int e = 0; e < 8; ++e) acc[t][e] = 0.0f;

    const float* __restrict__ ap = A + (size_t)(rowBase + nl) * lda;

    int ncl[4];
#pragma unroll
    for (int t = 0; t < 4; ++t) {
        int n = colBase + t * 16 + nl;
        ncl[t] = (n < N) ? n : (N - 1);  // clamp for load; store is masked
    }

    for (int kb = 0; kb < K; kb += 32) {
        // A fragment: contiguous K layout (lanes 0-15: K=kb+0..15,
        // lanes 16-31: K=kb+16..31), convert f32 -> f16 inline.
        v16h afr;
        const float* p0 = ap + kb + hi * 16;
        const float* p1 = p0 + 8;
#pragma unroll
        for (int e = 0; e < 8; ++e) {
            afr[e]     = (_Float16)p0[e];
            afr[e + 8] = (_Float16)p1[e];
        }
#pragma unroll
        for (int t = 0; t < 4; ++t) {
            // B fragment: lane nl holds W row n, 16 consecutive f16 (32B).
            const _Float16* wp = W + (size_t)ncl[t] * ldw + kb + hi * 16;
            v16h bfr = *(const v16h*)wp;
            acc[t] = wmma_f16(afr, bfr, acc[t]);
        }
    }

#pragma unroll
    for (int t = 0; t < 4; ++t) {
        int n = colBase + t * 16 + nl;
        if (n < N) {
            float bi = bias ? bias[n] : 0.0f;
#pragma unroll
            for (int v = 0; v < 8; ++v) {
                int m = rowBase + hi * 8 + v;   // C/D layout: elem v -> M row
                float val = acc[t][v] + bi;
                if (resid) val += resid[(size_t)m * ldr + n];
                if (relu)  val = fmaxf(val, 0.0f);
                C[(size_t)m * ldc + n] = val;
            }
        }
    }
}

// ---------------------------------------------------------------------------
// 16xN tile GEMM from LDS activations (fused attention inner GEMMs).
// sA: 16x256 fp32 in LDS, W: f16 [256,256] in global, sOut: 16x256 fp32 LDS.
// All 128 threads participate (EXEC all-ones around WMMA).
// ---------------------------------------------------------------------------
static __device__ void tile_gemm16(const float* __restrict__ sA,
                                   const _Float16* __restrict__ W,
                                   float* __restrict__ sOut, int relu)
{
    const int wave = threadIdx.x >> 5;
    const int lane = threadIdx.x & 31;
    const int nl   = lane & 15;
    const int hi   = lane >> 4;

    v8f acc[4];
#pragma unroll
    for (int t = 0; t < 4; ++t)
#pragma unroll
        for (int e = 0; e < 8; ++e) acc[t][e] = 0.0f;

    const float* ap = sA + nl * 256;
    for (int kb = 0; kb < 256; kb += 32) {
        v16h afr;
        const float* p0 = ap + kb + hi * 16;
        const float* p1 = p0 + 8;
#pragma unroll
        for (int e = 0; e < 8; ++e) {
            afr[e]     = (_Float16)p0[e];
            afr[e + 8] = (_Float16)p1[e];
        }
#pragma unroll
        for (int t = 0; t < 4; ++t) {
            int n = wave * 64 + t * 16 + nl;
            v16h bfr = *(const v16h*)(W + (size_t)n * 256 + kb + hi * 16);
            acc[t] = wmma_f16(afr, bfr, acc[t]);
        }
    }
#pragma unroll
    for (int t = 0; t < 4; ++t) {
        int n = wave * 64 + t * 16 + nl;
#pragma unroll
        for (int v = 0; v < 8; ++v) {
            float val = acc[t][v];
            if (relu) val = fmaxf(val, 0.0f);
            sOut[(v + hi * 8) * 256 + n] = val;
        }
    }
}

// ---------------------------------------------------------------------------
// Fused point-transformer attention: one block (128 thr) per query point.
// pos = relu(rel@pos1^T)@pos2^T ; t = q - k_nbr + pos ; h = relu(t@att1^T)
// a = h@att2^T ; attn = softmax_k(a/16) ; res = sum_k attn*(v_nbr+pos)
// ---------------------------------------------------------------------------
__global__ __launch_bounds__(128)
void attn_kernel(const float* __restrict__ xyz, const int* __restrict__ idx,
                 const float* __restrict__ q, const float* __restrict__ klin,
                 const float* __restrict__ vlin,
                 const float* __restrict__ pos1,
                 const _Float16* __restrict__ pos2h,
                 const _Float16* __restrict__ att1h,
                 const _Float16* __restrict__ att2h,
                 float* __restrict__ res)
{
    __shared__ float sPos[16 * 256];
    __shared__ float sT[16 * 256];
    __shared__ float sH[16 * 256];
    __shared__ float sQ[256];
    __shared__ float sRel[48];
    __shared__ int   sIdx[16];

    const int m   = blockIdx.x;       // b*128 + i
    const int b   = m >> 7;
    const int tid = threadIdx.x;

    if (tid < 16) sIdx[tid] = idx[m * 16 + tid];
    for (int c = tid; c < 256; c += 128) sQ[c] = q[(size_t)m * 256 + c];
    __syncthreads();

    if (tid < 48) {
        int kk = tid / 3, c = tid % 3;
        sRel[tid] = xyz[(size_t)m * 3 + c]
                  - xyz[((size_t)(b * NPTS) + sIdx[kk]) * 3 + c];
    }
    __syncthreads();

    // pos_h = relu(rel @ pos1^T)  (K=3, plain VALU)
    for (int e = tid; e < 4096; e += 128) {
        int kk = e >> 8, c = e & 255;
        float a = sRel[kk * 3 + 0] * pos1[c * 3 + 0]
                + sRel[kk * 3 + 1] * pos1[c * 3 + 1]
                + sRel[kk * 3 + 2] * pos1[c * 3 + 2];
        sH[e] = fmaxf(a, 0.0f);
    }
    __syncthreads();

    tile_gemm16(sH, pos2h, sPos, 0);          // pos
    __syncthreads();

    for (int e = tid; e < 4096; e += 128) {
        int kk = e >> 8, c = e & 255;
        sT[e] = sQ[c] - klin[((size_t)(b * NPTS) + sIdx[kk]) * 256 + c] + sPos[e];
    }
    __syncthreads();

    tile_gemm16(sT, att1h, sH, 1);            // h = relu(t @ att1^T)
    __syncthreads();
    tile_gemm16(sH, att2h, sT, 0);            // a = h @ att2^T
    __syncthreads();

    // softmax over neighbors (axis k) per channel, scale 1/sqrt(256)=1/16
    for (int c = tid; c < 256; c += 128) {
        float mx = -1e30f;
        for (int k = 0; k < 16; ++k) mx = fmaxf(mx, sT[k * 256 + c]);
        float sum = 0.0f;
        for (int k = 0; k < 16; ++k) {
            float e2 = expf((sT[k * 256 + c] - mx) * 0.0625f);
            sT[k * 256 + c] = e2;
            sum += e2;
        }
        float inv = 1.0f / sum;
        float acc = 0.0f;
        for (int k = 0; k < 16; ++k) {
            float vv = vlin[((size_t)(b * NPTS) + sIdx[k]) * 256 + c];
            acc += sT[k * 256 + c] * inv * (vv + sPos[k * 256 + c]);
        }
        res[(size_t)m * 256 + c] = acc;
    }
}

// ---------------------------------------------------------------------------
// LayerNorm over last dim (256), one row per 256-thread block, opt residual.
// ---------------------------------------------------------------------------
__global__ __launch_bounds__(256)
void ln_kernel(const float* __restrict__ in, float* __restrict__ out,
               const float* __restrict__ resid)
{
    __shared__ float r1[256];
    __shared__ float r2[256];
    const int row = blockIdx.x, t = threadIdx.x;
    float v = in[(size_t)row * 256 + t];
    r1[t] = v; r2[t] = v * v;
    __syncthreads();
    for (int off = 128; off > 0; off >>= 1) {
        if (t < off) { r1[t] += r1[t + off]; r2[t] += r2[t + off]; }
        __syncthreads();
    }
    float mean = r1[0] * (1.0f / 256.0f);
    float var  = r2[0] * (1.0f / 256.0f) - mean * mean;
    float o = (v - mean) * rsqrtf(var + 1e-5f);
    if (resid) o += resid[(size_t)row * 256 + t];
    out[(size_t)row * 256 + t] = o;
}

// ---------------------------------------------------------------------------
// KNN via stable rank counting. mode 0: order[::8] (global); 1: top-16 (local)
// ---------------------------------------------------------------------------
__global__ __launch_bounds__(128)
void knn_kernel(const float* __restrict__ xyz, int* __restrict__ idxo, int mode)
{
    __shared__ float sd[128];
    const int bi = blockIdx.x;           // b*128 + i
    const int b  = bi >> 7;
    const int j  = threadIdx.x;
    float dx = xyz[(size_t)bi * 3 + 0] - xyz[((size_t)(b * NPTS) + j) * 3 + 0];
    float dy = xyz[(size_t)bi * 3 + 1] - xyz[((size_t)(b * NPTS) + j) * 3 + 1];
    float dz = xyz[(size_t)bi * 3 + 2] - xyz[((size_t)(b * NPTS) + j) * 3 + 2];
    float d = dx * dx + dy * dy + dz * dz;
    sd[j] = d;
    __syncthreads();
    int rank = 0;
    for (int jj = 0; jj < 128; ++jj) {
        float o = sd[jj];
        rank += (o < d) || (o == d && jj < j);
    }
    if (mode == 1) {
        if (rank < 16) idxo[(size_t)bi * 16 + rank] = j;
    } else {
        if ((rank & 7) == 0) idxo[(size_t)bi * 16 + (rank >> 3)] = j;
    }
}

// --------------------------- elementwise helpers ---------------------------
__global__ void transpose_feat_kernel(const float* __restrict__ feat,
                                      float* __restrict__ f)
{
    int e = blockIdx.x * 256 + threadIdx.x;
    if (e >= BB * NPTS * DD) return;
    int b = e / (NPTS * DD);
    int r = e - b * NPTS * DD;
    int n = r / DD, c = r - n * DD;
    f[e] = feat[((size_t)b * DD + c) * NPTS + n];
}

__global__ void tohalf_kernel(const float* __restrict__ s,
                              _Float16* __restrict__ d, int n)
{
    int e = blockIdx.x * 256 + threadIdx.x;
    if (e < n) d[e] = (_Float16)s[e];
}

__global__ void tohalf_pad_kernel(const float* __restrict__ s,
                                  _Float16* __restrict__ d,
                                  int rows, int kin, int kout)
{
    int e = blockIdx.x * 256 + threadIdx.x;
    if (e >= rows * kout) return;
    int n = e / kout, c = e - n * kout;
    d[e] = (c < kin) ? (_Float16)s[(size_t)n * kin + c] : (_Float16)0.0f;
}

__global__ void build_actv_kernel(const float* __restrict__ xyz,
                                  const float* __restrict__ fgl,
                                  float* __restrict__ actv)
{
    int e = blockIdx.x * 256 + threadIdx.x;
    if (e >= M1 * 288) return;
    int m = e / 288, c = e - m * 288;
    float v;
    if (c < 3)        v = xyz[(size_t)m * 3 + c];
    else if (c < 259) v = fgl[(size_t)m * 256 + (c - 3)];
    else              v = 0.0f;
    actv[e] = v;
}

__global__ void split_vote_kernel(const float* __restrict__ vote,
                                  float* __restrict__ vxyz,
                                  float* __restrict__ vfeat)
{
    int e = blockIdx.x * 256 + threadIdx.x;
    if (e >= M1 * 259) return;
    int m = e / 259, c = e - m * 259;
    float v = vote[e];
    if (c < 3) vxyz[(size_t)m * 3 + c] = v;
    else       vfeat[(size_t)m * 256 + (c - 3)] = v;
}

// ---------------------------------------------------------------------------
// FPS: one block per batch, 128 threads, 63 sequential argmax steps.
// ---------------------------------------------------------------------------
__global__ __launch_bounds__(128)
void fps_kernel(const float* __restrict__ vxyz, int* __restrict__ cidx,
                float* __restrict__ centers)
{
    __shared__ float sv[128];
    __shared__ int   si[128];
    __shared__ int   sSel[64];
    const int b = blockIdx.x, t = threadIdx.x;
    float px = vxyz[((size_t)(b * NPTS) + t) * 3 + 0];
    float py = vxyz[((size_t)(b * NPTS) + t) * 3 + 1];
    float pz = vxyz[((size_t)(b * NPTS) + t) * 3 + 2];
    float dm = 1e10f;
    int last = 0;
    if (t == 0) sSel[0] = 0;
    for (int s = 1; s < NPROP; ++s) {
        float lx = vxyz[((size_t)(b * NPTS) + last) * 3 + 0];
        float ly = vxyz[((size_t)(b * NPTS) + last) * 3 + 1];
        float lz = vxyz[((size_t)(b * NPTS) + last) * 3 + 2];
        float dx = px - lx, dy = py - ly, dz = pz - lz;
        dm = fminf(dm, dx * dx + dy * dy + dz * dz);
        sv[t] = dm; si[t] = t;
        __syncthreads();
        for (int off = 64; off > 0; off >>= 1) {
            if (t < off) {
                if (sv[t + off] > sv[t]) { sv[t] = sv[t + off]; si[t] = si[t + off]; }
            }
            __syncthreads();
        }
        last = si[0];
        if (t == 0) sSel[s] = last;
        __syncthreads();
    }
    if (t < NPROP) {
        int c = sSel[t];
        cidx[(size_t)b * NPROP + t] = c;
        centers[((size_t)(b * NPROP) + t) * 3 + 0] = vxyz[((size_t)(b * NPTS) + c) * 3 + 0];
        centers[((size_t)(b * NPROP) + t) * 3 + 1] = vxyz[((size_t)(b * NPTS) + c) * 3 + 1];
        centers[((size_t)(b * NPROP) + t) * 3 + 2] = vxyz[((size_t)(b * NPTS) + c) * 3 + 2];
    }
}

__global__ __launch_bounds__(128)
void ballq_kernel(const float* __restrict__ vxyz,
                  const float* __restrict__ centers, int* __restrict__ gidx)
{
    __shared__ int sq[128];
    __shared__ int sg[16];
    const int bs = blockIdx.x;    // b*64+s
    const int b  = bs >> 6;
    const int t  = threadIdx.x;
    float cx = centers[(size_t)bs * 3 + 0];
    float cy = centers[(size_t)bs * 3 + 1];
    float cz = centers[(size_t)bs * 3 + 2];
    float dx = vxyz[((size_t)(b * NPTS) + t) * 3 + 0] - cx;
    float dy = vxyz[((size_t)(b * NPTS) + t) * 3 + 1] - cy;
    float dz = vxyz[((size_t)(b * NPTS) + t) * 3 + 2] - cz;
    int qual = (dx * dx + dy * dy + dz * dz < R2C) ? 1 : 0;
    sq[t] = qual;
    __syncthreads();
    int rank = 0, total = 0;
    for (int j = 0; j < 128; ++j) { total += sq[j]; if (j < t) rank += sq[j]; }
    if (qual && rank < NSAMP) sg[rank] = t;
    __syncthreads();
    if (t < NSAMP) {
        int val = (total > 0) ? ((t < total) ? sg[t] : sg[0]) : 0;
        gidx[(size_t)bs * NSAMP + t] = val;
    }
}

__global__ void build_gact_kernel(const float* __restrict__ vxyz,
                                  const float* __restrict__ centers,
                                  const float* __restrict__ vfeat,
                                  const int* __restrict__ gidx,
                                  float* __restrict__ gact)
{
    int e = blockIdx.x * 256 + threadIdx.x;
    if (e >= MSA * 288) return;
    int m = e / 288, c = e - m * 288;
    int b = m >> 10, r = m & 1023, s = r >> 4, kk = r & 15;
    int j = gidx[((size_t)(b * NPROP) + s) * NSAMP + kk];
    float v;
    if (c < 3)
        v = vxyz[((size_t)(b * NPTS) + j) * 3 + c]
          - centers[((size_t)(b * NPROP) + s) * 3 + c];
    else if (c < 259)
        v = vfeat[((size_t)(b * NPTS) + j) * 256 + (c - 3)];
    else
        v = 0.0f;
    gact[e] = v;
}

__global__ void maxpool_kernel(const float* __restrict__ g,
                               float* __restrict__ pf)
{
    int e = blockIdx.x * 256 + threadIdx.x;
    if (e >= MP * 256) return;
    int r = e / 256, c = e - r * 256;
    float mx = -1e30f;
    for (int k = 0; k < NSAMP; ++k)
        mx = fmaxf(mx, g[((size_t)(r * NSAMP) + k) * 256 + c]);
    pf[e] = mx;
}

__global__ void imp_kernel(const float* __restrict__ h,
                           const float* __restrict__ w,
                           const float* __restrict__ bias,
                           float* __restrict__ out)
{
    int m = blockIdx.x * 256 + threadIdx.x;
    if (m >= M1) return;
    float acc = 0.0f;
    for (int c = 0; c < 256; ++c) acc += h[(size_t)m * 256 + c] * w[c];
    out[m] = acc + bias[0];
}

__global__ void boxes_kernel(const float* __restrict__ h_off,
                             const float* __restrict__ h_ang,
                             const float* __restrict__ h_sc,
                             const float* __restrict__ off2_w,
                             const float* __restrict__ off2_b,
                             const float* __restrict__ ang2_w,
                             const float* __restrict__ ang2_b,
                             const float* __restrict__ sc2_w,
                             const float* __restrict__ sc2_b,
                             const float* __restrict__ centers,
                             float* __restrict__ boxes)
{
    int gid = blockIdx.x * 256 + threadIdx.x;
    if (gid >= MP * 5) return;
    int m = gid / 5, j = gid - m * 5;
    const float* h; const float* w; float bi; float extra = 0.0f;
    if (j < 3)      { h = h_off + (size_t)m * 256; w = off2_w + j * 256;
                      bi = off2_b[j]; extra = centers[(size_t)m * 3 + j]; }
    else if (j == 3){ h = h_ang + (size_t)m * 256; w = ang2_w; bi = ang2_b[0]; }
    else            { h = h_sc  + (size_t)m * 256; w = sc2_w;  bi = sc2_b[0]; }
    float acc = 0.0f;
    for (int c = 0; c < 256; ++c) acc += h[c] * w[c];
    boxes[gid] = acc + bi + extra;
}

// ---------------------------------------------------------------------------
// Host-side orchestration
// ---------------------------------------------------------------------------
extern "C" void kernel_launch(void* const* d_in, const int* in_sizes, int n_in,
                              void* d_out, int out_size, void* d_ws, size_t ws_size,
                              hipStream_t stream)
{
    (void)in_sizes; (void)n_in; (void)out_size; (void)ws_size;

    // jax pytree flattening orders dict keys alphabetically.
    const float* xyz     = (const float*)d_in[0];
    const float* feature = (const float*)d_in[1];
    const float* ang1    = (const float*)d_in[2];
    const float* ang2_b  = (const float*)d_in[3];
    const float* ang2_w  = (const float*)d_in[4];
    // gt: att1 att2 pos1 pos2 post_b post_w pre_b pre_w proj wk wq wv
    const float* gt_att1 = (const float*)d_in[5];
    const float* gt_att2 = (const float*)d_in[6];
    const float* gt_pos1 = (const float*)d_in[7];
    const float* gt_pos2 = (const float*)d_in[8];
    const float* gt_post_b = (const float*)d_in[9];
    const float* gt_post_w = (const float*)d_in[10];
    const float* gt_pre_b  = (const float*)d_in[11];
    const float* gt_pre_w  = (const float*)d_in[12];
    const float* gt_proj   = (const float*)d_in[13];
    const float* gt_wk = (const float*)d_in[14];
    const float* gt_wq = (const float*)d_in[15];
    const float* gt_wv = (const float*)d_in[16];
    const float* imp1   = (const float*)d_in[17];
    const float* imp2_b = (const float*)d_in[18];
    const float* imp2_w = (const float*)d_in[19];
    const float* lt_att1 = (const float*)d_in[20];
    const float* lt_att2 = (const float*)d_in[21];
    const float* lt_pos1 = (const float*)d_in[22];
    const float* lt_pos2 = (const float*)d_in[23];
    const float* lt_post_b = (const float*)d_in[24];
    const float* lt_post_w = (const float*)d_in[25];
    const float* lt_pre_b  = (const float*)d_in[26];
    const float* lt_pre_w  = (const float*)d_in[27];
    const float* lt_proj   = (const float*)d_in[28];
    const float* lt_wk = (const float*)d_in[29];
    const float* lt_wq = (const float*)d_in[30];
    const float* lt_wv = (const float*)d_in[31];
    const float* off1   = (const float*)d_in[32];
    const float* off2_b = (const float*)d_in[33];
    const float* off2_w = (const float*)d_in[34];
    const float* sa1 = (const float*)d_in[35];
    const float* sa2 = (const float*)d_in[36];
    const float* sa3 = (const float*)d_in[37];
    const float* sc1   = (const float*)d_in[38];
    const float* sc2_b = (const float*)d_in[39];
    const float* sc2_w = (const float*)d_in[40];
    const float* vote1   = (const float*)d_in[41];
    const float* vote2_b = (const float*)d_in[42];
    const float* vote2_w = (const float*)d_in[43];

    // Output regions (tuple concat: boxes, vote_xyz, centers, imp)
    float* out = (float*)d_out;
    float* boxes_o = out;                               // 128*64*5
    float* vxyz_o  = out + (size_t)BB * NPROP * 5;      // 128*128*3
    float* cent_o  = vxyz_o + (size_t)BB * NPTS * 3;    // 128*64*3
    float* imp_o   = cent_o + (size_t)BB * NPROP * 3;   // 128*128

    // Workspace carving (256B-aligned regions)
    char* base = (char*)d_ws;
    size_t cur = 0;
    auto allocF = [&](size_t n) -> float* {
        float* p = (float*)(base + cur);
        cur = (cur + n * sizeof(float) + 255) & ~(size_t)255; return p; };
    auto allocH = [&](size_t n) -> _Float16* {
        _Float16* p = (_Float16*)(base + cur);
        cur = (cur + n * sizeof(_Float16) + 255) & ~(size_t)255; return p; };
    auto allocI = [&](size_t n) -> int* {
        int* p = (int*)(base + cur);
        cur = (cur + n * sizeof(int) + 255) & ~(size_t)255; return p; };

    float* bF   = allocF((size_t)M1 * 256);   // feats (transposed input)
    float* bX   = allocF((size_t)M1 * 256);
    float* bQ   = allocF((size_t)M1 * 256);
    float* bK   = allocF((size_t)M1 * 256);
    float* bV   = allocF((size_t)M1 * 256);
    float* bRes = allocF((size_t)M1 * 256);
    float* bT1  = allocF((size_t)M1 * 256);
    float* bFg  = allocF((size_t)M1 * 256);
    float* bFgl = allocF((size_t)M1 * 256);
    float* actv = allocF((size_t)M1 * 288);
    float* vote = allocF((size_t)M1 * 259);
    float* vfeat= allocF((size_t)M1 * 256);
    float* gact = allocF((size_t)MSA * 288);  // also reused as g2
    float* g1   = allocF((size_t)MSA * 256);
    int*   idxg = allocI((size_t)M1 * 16);
    int*   idxl = allocI((size_t)M1 * 16);
    int*   cidx = allocI((size_t)BB * NPROP);
    int*   gidx = allocI((size_t)BB * NPROP * NSAMP);

    // f16 weight copies
    _Float16 *h_gt[9], *h_lt[9];
    const float* gt_src[9] = {gt_pre_w, gt_wq, gt_wk, gt_wv, gt_pos2,
                              gt_att1, gt_att2, gt_proj, gt_post_w};
    const float* lt_src[9] = {lt_pre_w, lt_wq, lt_wk, lt_wv, lt_pos2,
                              lt_att1, lt_att2, lt_proj, lt_post_w};
    for (int i = 0; i < 9; ++i) h_gt[i] = allocH(256 * 256);
    for (int i = 0; i < 9; ++i) h_lt[i] = allocH(256 * 256);
    _Float16* h_imp1  = allocH(256 * 256);
    _Float16* h_vote1 = allocH(256 * 288);
    _Float16* h_vote2 = allocH(259 * 256);
    _Float16* h_sa1   = allocH(256 * 288);
    _Float16* h_sa2   = allocH(256 * 256);
    _Float16* h_sa3   = allocH(256 * 256);
    _Float16* h_off1  = allocH(256 * 256);
    _Float16* h_ang1  = allocH(256 * 256);
    _Float16* h_sc1   = allocH(256 * 256);

    auto cvt = [&](const float* s, _Float16* d, int n) {
        tohalf_kernel<<<(n + 255) / 256, 256, 0, stream>>>(s, d, n); };
    for (int i = 0; i < 9; ++i) cvt(gt_src[i], h_gt[i], 256 * 256);
    for (int i = 0; i < 9; ++i) cvt(lt_src[i], h_lt[i], 256 * 256);
    cvt(imp1,  h_imp1, 256 * 256);
    cvt(vote2_w, h_vote2, 259 * 256);
    cvt(sa2, h_sa2, 256 * 256);  cvt(sa3, h_sa3, 256 * 256);
    cvt(off1, h_off1, 256 * 256); cvt(ang1, h_ang1, 256 * 256);
    cvt(sc1, h_sc1, 256 * 256);
    tohalf_pad_kernel<<<(256 * 288 + 255) / 256, 256, 0, stream>>>(vote1, h_vote1, 256, 259, 288);
    tohalf_pad_kernel<<<(256 * 288 + 255) / 256, 256, 0, stream>>>(sa1,   h_sa1,   256, 259, 288);

    auto gemm = [&](const float* A, int lda, const _Float16* W, int ldw,
                    float* C, int ldc, const float* bias,
                    const float* resid, int ldr, int M, int N, int K, int relu) {
        dim3 g(M / 64, (N + 63) / 64);
        gemm_wmma_kernel<<<g, 128, 0, stream>>>(A, lda, W, ldw, C, ldc,
                                                bias, resid, ldr, N, K, relu);
    };
    auto ln = [&](const float* in, float* o, const float* resid) {
        ln_kernel<<<M1, 256, 0, stream>>>(in, o, resid); };

    // ---- input transpose + KNN ----
    transpose_feat_kernel<<<(M1 * 256 + 255) / 256, 256, 0, stream>>>(feature, bF);
    knn_kernel<<<M1, 128, 0, stream>>>(xyz, idxg, 0);
    knn_kernel<<<M1, 128, 0, stream>>>(xyz, idxl, 1);

    // ---- transformer block (shared routine) ----
    auto run_block = [&](const float* fin, const int* idx, _Float16* const* h,
                         const float* pre_b, const float* pos1,
                         const float* post_b, float* fout) {
        gemm(fin, 256, h[0], 256, bX, 256, pre_b, nullptr, 0, M1, 256, 256, 0);
        gemm(bX, 256, h[1], 256, bQ, 256, nullptr, nullptr, 0, M1, 256, 256, 0);
        ln(bQ, bQ, nullptr);
        gemm(bX, 256, h[2], 256, bK, 256, nullptr, nullptr, 0, M1, 256, 256, 0);
        ln(bK, bK, nullptr);
        gemm(bX, 256, h[3], 256, bV, 256, nullptr, nullptr, 0, M1, 256, 256, 0);
        ln(bV, bV, nullptr);
        attn_kernel<<<M1, 128, 0, stream>>>(xyz, idx, bQ, bK, bV, pos1,
                                            h[4], h[5], h[6], bRes);
        gemm(bRes, 256, h[7], 256, bT1, 256, nullptr, nullptr, 0, M1, 256, 256, 0);
        ln(bT1, bT1, nullptr);
        gemm(bT1, 256, h[8], 256, bRes, 256, post_b, nullptr, 0, M1, 256, 256, 0);
        ln(bRes, fout, fin);
    };

    run_block(bF, idxg, h_gt, gt_pre_b, gt_pos1, gt_post_b, bFg);   // global

    // ---- importance head (uses fg) ----
    gemm(bFg, 256, h_imp1, 256, bT1, 256, nullptr, nullptr, 0, M1, 256, 256, 1);
    imp_kernel<<<(M1 + 255) / 256, 256, 0, stream>>>(bT1, imp2_w, imp2_b, imp_o);

    run_block(bFg, idxl, h_lt, lt_pre_b, lt_pos1, lt_post_b, bFgl); // local

    // ---- voting ----
    build_actv_kernel<<<(M1 * 288 + 255) / 256, 256, 0, stream>>>(xyz, bFgl, actv);
    gemm(actv, 288, h_vote1, 288, bX, 256, nullptr, nullptr, 0, M1, 256, 288, 1);
    gemm(bX, 256, h_vote2, 256, vote, 259, vote2_b, actv, 288, M1, 259, 256, 0);
    split_vote_kernel<<<(M1 * 259 + 255) / 256, 256, 0, stream>>>(vote, vxyz_o, vfeat);

    // ---- FPS, ball query, grouping ----
    fps_kernel<<<BB, 128, 0, stream>>>(vxyz_o, cidx, cent_o);
    ballq_kernel<<<BB * NPROP, 128, 0, stream>>>(vxyz_o, cent_o, gidx);
    build_gact_kernel<<<(MSA * 288 + 255) / 256, 256, 0, stream>>>(vxyz_o, cent_o,
                                                                   vfeat, gidx, gact);

    // ---- SA MLP (sa1->sa2->sa3 with relu), g2 reuses gact region ----
    gemm(gact, 288, h_sa1, 288, g1, 256, nullptr, nullptr, 0, MSA, 256, 288, 1);
    gemm(g1, 256, h_sa2, 256, gact, 256, nullptr, nullptr, 0, MSA, 256, 256, 1);
    gemm(gact, 256, h_sa3, 256, g1, 256, nullptr, nullptr, 0, MSA, 256, 256, 1);
    maxpool_kernel<<<(MP * 256 + 255) / 256, 256, 0, stream>>>(g1, bK);  // pf in bK

    // ---- heads (reuse dead transformer buffers) ----
    gemm(bK, 256, h_off1, 256, bV, 256, nullptr, nullptr, 0, MP, 256, 256, 1);
    gemm(bK, 256, h_ang1, 256, bRes, 256, nullptr, nullptr, 0, MP, 256, 256, 1);
    gemm(bK, 256, h_sc1, 256, bT1, 256, nullptr, nullptr, 0, MP, 256, 256, 1);
    boxes_kernel<<<(MP * 5 + 255) / 256, 256, 0, stream>>>(
        bV, bRes, bT1, off2_w, off2_b, ang2_w, ang2_b, sc2_w, sc2_b,
        cent_o, boxes_o);
}
